// TransformerBlock_20031727469026
// MI455X (gfx1250) — matrix operand order repaired and checked by reading, not compiled, not run on hardware
//
#include <hip/hip_runtime.h>
#include <hip/hip_bf16.h>
#include <math.h>

// ---------------------------------------------------------------------------
// Types for CDNA5 WMMA (wave32): v_wmma_f32_16x16x32_bf16
// ---------------------------------------------------------------------------
typedef __bf16 bf16;
typedef __attribute__((ext_vector_type(16))) __bf16 v16bf;
typedef __attribute__((ext_vector_type(8)))  __bf16 v8bf;
typedef __attribute__((ext_vector_type(8)))  float  v8f;

static __device__ inline v16bf vb_zero16() {
  v16bf r;
#pragma unroll
  for (int i = 0; i < 16; ++i) r[i] = (bf16)0.0f;
  return r;
}

static __device__ inline v8f vf_zero8() {
  v8f r;
#pragma unroll
  for (int i = 0; i < 8; ++i) r[i] = 0.0f;
  return r;
}

static __device__ inline v8f wmma_bf16(v16bf a, v16bf b, v8f c) {
  // (neg_a, A, neg_b, B, c_mod, C, reuse_a, reuse_b)
  return __builtin_amdgcn_wmma_f32_16x16x32_bf16(false, a, false, b, (short)0, c,
                                                 false, false);
}

// Async copy 16B global -> LDS (CDNA5, tracked by ASYNCcnt).
static __device__ inline void async_ld_b128(unsigned lds_addr, const bf16* g) {
  unsigned long long ga = (unsigned long long)g;
  asm volatile("global_load_async_to_lds_b128 %0, %1, off"
               :: "v"(lds_addr), "v"(ga) : "memory");
}

#define PITCH 40  // LDS tile pitch in bf16 elems: 80B rows -> conflict-free b128

// ---------------------------------------------------------------------------
// Fragment loaders (A/B in contiguous k order, same for both operands; C/D per CDNA5 ISA 7.12.2)
// A (16x32 MxK): lane L holds row M=L&15; lane<16 elems -> K 0..15,
//   lane>=16 -> K 16..31.
// B (32x16 KxN): lane n<16 holds col n, K k0..k0+15; lane>=16 col n-16,
//   K k0+16..k0+31.
// C/D (16x16 f32): vgpr r -> (M=r, N=lane) lanes 0..15, (M=r+8, N=lane-16).
// ---------------------------------------------------------------------------
static __device__ inline v16bf load_a_frag(const bf16* __restrict__ A, int lda,
                                           int row0, int k0) {
  int lane = threadIdx.x & 31;
  int row  = row0 + (lane & 15);
  int kb   = k0 + ((lane >> 4) << 4);  // +0 or +16
  const bf16* p = A + (size_t)row * lda + kb;
  v8bf lo = *(const v8bf*)(p);
  v8bf hi = *(const v8bf*)(p + 8);
  v16bf r;
#pragma unroll
  for (int i = 0; i < 8; ++i) { r[i] = lo[i]; r[i + 8] = hi[i]; }
  return r;
}

// A fragment from LDS tile laid out [rows][PITCH].
static __device__ inline v16bf load_a_frag_lds(const bf16* as, int row0loc) {
  int lane = threadIdx.x & 31;
  int row = row0loc + (lane & 15);
  int kb = (lane >> 4) << 4;  // +0 or +16
  const bf16* p = as + row * PITCH + kb;
  v8bf lo = *(const v8bf*)(p);
  v8bf hi = *(const v8bf*)(p + 8);
  v16bf r;
#pragma unroll
  for (int i = 0; i < 8; ++i) { r[i] = lo[i]; r[i + 8] = hi[i]; }
  return r;
}

// Bt (global) stored transposed: Bt[n][k], ldb = K.
static __device__ inline v16bf load_bt_frag(const bf16* __restrict__ Bt, int ldb,
                                            int col0, int k0) {
  int lane = threadIdx.x & 31;
  int col  = col0 + (lane & 15);
  int kb   = k0 + ((lane >> 4) << 4);  // +0 or +16
  const bf16* p = Bt + (size_t)col * ldb + kb;
  v8bf lo = *(const v8bf*)(p);
  v8bf hi = *(const v8bf*)(p + 8);
  v16bf r;
#pragma unroll
  for (int i = 0; i < 8; ++i) { r[i] = lo[i]; r[i + 8] = hi[i]; }
  return r;
}

// B fragment from LDS tile laid out [64 cols][PITCH].
static __device__ inline v16bf load_b_frag_lds(const bf16* bs, int col0loc) {
  int lane = threadIdx.x & 31;
  int col = col0loc + (lane & 15);
  int kb = (lane >> 4) << 4;  // +0 or +16
  const bf16* p = bs + col * PITCH + kb;
  v8bf lo = *(const v8bf*)(p);
  v8bf hi = *(const v8bf*)(p + 8);
  v16bf r;
#pragma unroll
  for (int i = 0; i < 8; ++i) { r[i] = lo[i]; r[i + 8] = hi[i]; }
  return r;
}

// ---------------------------------------------------------------------------
// LayerNorm: one block (256 thr) per row of 1024; fp32 in -> bf16 out.
// ---------------------------------------------------------------------------
__global__ __launch_bounds__(256) void ln_kernel(const float* __restrict__ x,
                                                 const float* __restrict__ w,
                                                 const float* __restrict__ b,
                                                 bf16* __restrict__ y, int D) {
  __shared__ float sh[8];
  int row = blockIdx.x;
  int tid = threadIdx.x;
  const float* xr = x + (size_t)row * D;
  float v[4];
  float s = 0.f;
#pragma unroll
  for (int i = 0; i < 4; ++i) { v[i] = xr[tid + 256 * i]; s += v[i]; }
#pragma unroll
  for (int m = 16; m >= 1; m >>= 1) s += __shfl_xor(s, m, 32);
  if ((tid & 31) == 0) sh[tid >> 5] = s;
  __syncthreads();
  float tot = 0.f;
#pragma unroll
  for (int i = 0; i < 8; ++i) tot += sh[i];
  float mean = tot * (1.0f / 1024.0f);
  __syncthreads();
  float vs = 0.f;
#pragma unroll
  for (int i = 0; i < 4; ++i) { float d = v[i] - mean; vs += d * d; }
#pragma unroll
  for (int m = 16; m >= 1; m >>= 1) vs += __shfl_xor(vs, m, 32);
  if ((tid & 31) == 0) sh[tid >> 5] = vs;
  __syncthreads();
  float vtot = 0.f;
#pragma unroll
  for (int i = 0; i < 8; ++i) vtot += sh[i];
  float rstd = rsqrtf(vtot * (1.0f / 1024.0f) + 1e-5f);
#pragma unroll
  for (int i = 0; i < 4; ++i) {
    int c = tid + 256 * i;
    y[(size_t)row * D + c] = (bf16)((v[i] - mean) * rstd * w[c] + b[c]);
  }
}

// ---------------------------------------------------------------------------
// Transpose + fp32->bf16 convert: W[K][N] -> Wt[N][K]. Block (32,8).
// ---------------------------------------------------------------------------
__global__ void transpose_f32_bf16(const float* __restrict__ W,
                                   bf16* __restrict__ Wt, int K, int N) {
  __shared__ float tile[32][33];
  int n0 = blockIdx.x * 32, k0 = blockIdx.y * 32;
  int tx = threadIdx.x, ty = threadIdx.y;
#pragma unroll
  for (int i = 0; i < 32; i += 8)
    tile[ty + i][tx] = W[(size_t)(k0 + ty + i) * N + n0 + tx];
  __syncthreads();
#pragma unroll
  for (int i = 0; i < 32; i += 8)
    Wt[(size_t)(n0 + ty + i) * K + k0 + tx] = (bf16)tile[tx][ty + i];
}

// W_Q/K/V: [16][1024][64] -> Wt[(n*64+h)][1024]. Block (32,8), grid (2,32,16).
__global__ void transpose_head_w(const float* __restrict__ W,
                                 bf16* __restrict__ Wt) {
  __shared__ float tile[32][33];
  int n = blockIdx.z;
  int h0 = blockIdx.x * 32, d0 = blockIdx.y * 32;
  int tx = threadIdx.x, ty = threadIdx.y;
#pragma unroll
  for (int i = 0; i < 32; i += 8)
    tile[ty + i][tx] = W[(size_t)n * 65536 + (size_t)(d0 + ty + i) * 64 + h0 + tx];
  __syncthreads();
#pragma unroll
  for (int i = 0; i < 32; i += 8)
    Wt[(size_t)(n * 64 + h0 + ty + i) * 1024 + d0 + tx] = (bf16)tile[tx][ty + i];
}

// ---------------------------------------------------------------------------
// WMMA GEMM: C[M,N] = A[M,K] (bf16) x Bt[N,K]^T (bf16) + bias
// Block = 8 waves; wave tile 32x64 -> block tile 256x64. BOTH A (256x32) and
// B (64x32) K-step tiles are staged global->LDS with CDNA5 async copies
// (ASYNCcnt), double-buffered so the inner loop is pure ds_load + WMMA.
// Epilogue MODE (compile-time):
//   0: bf16 store           1: bf16 store with exact GELU
//   2: fp32 store + residual 3: bf16 store V-transposed ([b,nh,s])
// ---------------------------------------------------------------------------
template <int MODE>
__global__ __launch_bounds__(256) void gemm_kernel(
    const bf16* __restrict__ A, const bf16* __restrict__ Bt,
    const float* __restrict__ bias, const float* __restrict__ resid,
    void* __restrict__ Cout, int M, int N, int K) {
  __shared__ bf16 asmem[2][256 * PITCH];  // 2 x 20KB
  __shared__ bf16 bsmem[2][64 * PITCH];   // 2 x 5KB

  int tid  = threadIdx.x;
  int wave = tid >> 5;
  int lane = tid & 31;
  int rowblk = blockIdx.y * 256;
  int row0loc = wave * 32;
  int col0 = blockIdx.x * 64;

  // Async-copy assignment (16B chunks):
  //   B: thread t -> col t>>2, chunk (t&3)*8        (64x32 tile, 1 op/thread)
  //   A: thread t -> rows (t>>2)+{0,64,128,192}, chunk (t&3)*8 (4 ops/thread)
  int arow = tid >> 2;
  int achk = (tid & 3) * 8;
  const bf16* gA = A + (size_t)(rowblk + arow) * K + achk;
  const bf16* gB = Bt + (size_t)(col0 + arow) * K + achk;  // arow<64 cols for B
  unsigned ldsA[2], ldsB[2];
#pragma unroll
  for (int u = 0; u < 2; ++u) {
    ldsA[u] = (unsigned)(uintptr_t)&asmem[u][arow * PITCH + achk];
    ldsB[u] = (unsigned)(uintptr_t)&bsmem[u][arow * PITCH + achk];
  }

  v8f acc[2][4];
#pragma unroll
  for (int rr = 0; rr < 2; ++rr)
#pragma unroll
    for (int c = 0; c < 4; ++c) acc[rr][c] = vf_zero8();

  // Prologue: fetch first A+B tiles into buffer 0.
#pragma unroll
  for (int i = 0; i < 4; ++i)
    async_ld_b128(ldsA[0] + i * 64 * PITCH * 2, gA + (size_t)(i * 64) * K);
  async_ld_b128(ldsB[0], gB);

  for (int k0 = 0; k0 < K; k0 += 32) {
    int buf = (k0 >> 5) & 1;
    asm volatile("s_wait_asynccnt 0" ::: "memory");
    __syncthreads();
    if (k0 + 32 < K) {
      int nb = buf ^ 1;
#pragma unroll
      for (int i = 0; i < 4; ++i)
        async_ld_b128(ldsA[nb] + i * 64 * PITCH * 2,
                      gA + (size_t)(i * 64) * K + k0 + 32);
      async_ld_b128(ldsB[nb], gB + k0 + 32);
    }

    const bf16* as = &asmem[buf][0];
    const bf16* bs = &bsmem[buf][0];
    v16bf a0 = load_a_frag_lds(as, row0loc);
    v16bf a1 = load_a_frag_lds(as, row0loc + 16);
#pragma unroll
    for (int c = 0; c < 4; ++c) {
      v16bf b = load_b_frag_lds(bs, c * 16);
      acc[0][c] = wmma_bf16(a0, b, acc[0][c]);
      acc[1][c] = wmma_bf16(a1, b, acc[1][c]);
    }
  }

  int hi8 = (lane >> 4) << 3;
#pragma unroll
  for (int rr = 0; rr < 2; ++rr) {
#pragma unroll
    for (int c = 0; c < 4; ++c) {
      int col = col0 + 16 * c + (lane & 15);
      float bv = bias[col];
#pragma unroll
      for (int r = 0; r < 8; ++r) {
        int m = rowblk + row0loc + rr * 16 + r + hi8;
        float val = acc[rr][c][r] + bv;
        if (MODE == 0) {
          ((bf16*)Cout)[(size_t)m * N + col] = (bf16)val;
        } else if (MODE == 1) {
          float g = 0.5f * val * (1.0f + erff(val * 0.70710678118654752f));
          ((bf16*)Cout)[(size_t)m * N + col] = (bf16)g;
        } else if (MODE == 2) {
          ((float*)Cout)[(size_t)m * N + col] = val + resid[(size_t)m * N + col];
        } else {  // MODE 3: store transposed for attention V: Vt[b*1024+col][s]
          int bb = m >> 11, s = m & 2047;
          ((bf16*)Cout)[((size_t)(bb * 1024 + col)) * 2048 + s] = (bf16)val;
        }
      }
    }
  }
}

// ---------------------------------------------------------------------------
// Flash-style causal attention. grid (S/128, N_HEADS, B), block 256 = 8 waves.
// Each wave owns a 16-row Q strip; iterates 16-key blocks with online softmax.
// q,k: [B,S,1024] (col = n*64+h). vt: [B,1024(nh),S]. z out: [B,S,1024].
// ---------------------------------------------------------------------------
__global__ __launch_bounds__(256) void attn_kernel(const bf16* __restrict__ q,
                                                   const bf16* __restrict__ k,
                                                   const bf16* __restrict__ vt,
                                                   bf16* __restrict__ z) {
  const int S = 2048, NH = 1024;
  __shared__ bf16 Plds[8][16][24];  // per-wave P staging, padded pitch 24

  int wave = threadIdx.x >> 5;
  int lane = threadIdx.x & 31;
  int hi8  = (lane >> 4) << 3;
  int b = blockIdx.z, n = blockIdx.y;
  int qtile = blockIdx.x * 128 + wave * 16;

  const bf16* qb = q + (size_t)b * S * NH + n * 64;   // row stride NH, K over h
  const bf16* kb_base = k + (size_t)b * S * NH + n * 64;
  const bf16* vb = vt + ((size_t)b * NH + n * 64) * S;  // row h, stride S
  bf16* Pw = &Plds[wave][0][0];

  // Q fragments (16x64 split into two 16x32 A-frags), kept in registers.
  v16bf qf0 = load_a_frag(qb, NH, qtile, 0);
  v16bf qf1 = load_a_frag(qb, NH, qtile, 32);

  float mrow[8], lrow[8];
#pragma unroll
  for (int r = 0; r < 8; ++r) { mrow[r] = -1.0e30f; lrow[r] = 0.0f; }
  v8f acc[4];
#pragma unroll
  for (int c = 0; c < 4; ++c) acc[c] = vf_zero8();

  for (int kt = 0; kt <= qtile; kt += 16) {
    // scores S = Q (16x64) x K^T (64x16); B-operand columns are K rows.
    v8f sc = vf_zero8();
    v16bf kf0 = load_bt_frag(kb_base, NH, kt, 0);
    sc = wmma_bf16(qf0, kf0, sc);
    v16bf kf1 = load_bt_frag(kb_base, NH, kt, 32);
    sc = wmma_bf16(qf1, kf1, sc);

    int keyc = kt + (lane & 15);
#pragma unroll
    for (int r = 0; r < 8; ++r) {
      int qrow = qtile + r + hi8;
      float sv = sc[r] * 0.125f;  // 1/sqrt(64)
      if (keyc > qrow) sv = -1.0e30f;
      sc[r] = sv;
    }
    // row max across the 16 lanes that hold each row
    float rm[8], ps[8], rs[8], alpha[8];
#pragma unroll
    for (int r = 0; r < 8; ++r) rm[r] = sc[r];
#pragma unroll
    for (int m = 8; m >= 1; m >>= 1)
#pragma unroll
      for (int r = 0; r < 8; ++r) rm[r] = fmaxf(rm[r], __shfl_xor(rm[r], m, 32));
#pragma unroll
    for (int r = 0; r < 8; ++r) {
      float mn = fmaxf(mrow[r], rm[r]);
      alpha[r] = __expf(mrow[r] - mn);
      mrow[r] = mn;
      ps[r] = __expf(sc[r] - mn);
      rs[r] = ps[r];
    }
#pragma unroll
    for (int m = 8; m >= 1; m >>= 1)
#pragma unroll
      for (int r = 0; r < 8; ++r) rs[r] += __shfl_xor(rs[r], m, 32);
#pragma unroll
    for (int r = 0; r < 8; ++r) lrow[r] = lrow[r] * alpha[r] + rs[r];

    // Stage P (C-layout) to LDS, reload in A-layout (upper K half zero).
#pragma unroll
    for (int r = 0; r < 8; ++r)
      Pw[(r + hi8) * 24 + (lane & 15)] = (bf16)ps[r];
    asm volatile("s_wait_dscnt 0" ::: "memory");
    v16bf pf;
    {
      int row = lane & 15;
      pf = vb_zero16();
      if (lane < 16) {
        v8bf lo = *(const v8bf*)(Pw + row * 24);
        v8bf hi = *(const v8bf*)(Pw + row * 24 + 8);
#pragma unroll
        for (int i = 0; i < 8; ++i) { pf[i] = lo[i]; pf[i + 8] = hi[i]; }
      }
    }

    // acc = acc*alpha + P (16x32, keys padded) x V (32x64, keys x h)
#pragma unroll
    for (int c = 0; c < 4; ++c) {
      v16bf vf = vb_zero16();
      if (lane < 16) {
        const bf16* p = vb + (size_t)(c * 16 + lane) * S + kt;
        v8bf lo = *(const v8bf*)(p);
        v8bf hi = *(const v8bf*)(p + 8);
#pragma unroll
        for (int i = 0; i < 8; ++i) { vf[i] = lo[i]; vf[i + 8] = hi[i]; }
      }
#pragma unroll
      for (int r = 0; r < 8; ++r) acc[c][r] *= alpha[r];
      acc[c] = wmma_bf16(pf, vf, acc[c]);
    }
  }

  // Normalize and store z[b, s, n*64+h]
#pragma unroll
  for (int c = 0; c < 4; ++c) {
#pragma unroll
    for (int r = 0; r < 8; ++r) {
      int s = qtile + r + hi8;
      int col = n * 64 + c * 16 + (lane & 15);
      z[((size_t)b * S + s) * NH + col] = (bf16)(acc[c][r] / lrow[r]);
    }
  }
}

// ---------------------------------------------------------------------------
// Host-side orchestration
// ---------------------------------------------------------------------------
extern "C" void kernel_launch(void* const* d_in, const int* in_sizes, int n_in,
                              void* d_out, int out_size, void* d_ws,
                              size_t ws_size, hipStream_t stream) {
  (void)in_sizes; (void)n_in; (void)out_size; (void)ws_size;
  const float* resid_pre = (const float*)d_in[0];
  const float* W_Q   = (const float*)d_in[1];
  const float* b_Q   = (const float*)d_in[2];
  const float* W_K   = (const float*)d_in[3];
  const float* b_K   = (const float*)d_in[4];
  const float* W_V   = (const float*)d_in[5];
  const float* b_V   = (const float*)d_in[6];
  const float* W_O   = (const float*)d_in[7];
  const float* b_O   = (const float*)d_in[8];
  const float* ln1_w = (const float*)d_in[9];
  const float* ln1_b = (const float*)d_in[10];
  const float* ln2_w = (const float*)d_in[11];
  const float* ln2_b = (const float*)d_in[12];
  const float* W_in  = (const float*)d_in[13];
  const float* b_in  = (const float*)d_in[14];
  const float* W_out = (const float*)d_in[15];
  const float* b_out = (const float*)d_in[16];
  float* out = (float*)d_out;

  const int M = 4096;   // B*S
  const int D = 1024;   // d_model
  const int F = 4096;   // d_mlp

  char* ws = (char*)d_ws;
  size_t off = 0;
  auto alloc = [&](size_t bytes) -> void* {
    void* p = ws + off;
    off += (bytes + 255) & ~(size_t)255;
    return p;
  };
  bf16* xn1    = (bf16*)alloc((size_t)M * D * 2);
  bf16* wq_t   = (bf16*)alloc((size_t)D * D * 2);
  bf16* wk_t   = (bf16*)alloc((size_t)D * D * 2);
  bf16* wv_t   = (bf16*)alloc((size_t)D * D * 2);
  bf16* wo_t   = (bf16*)alloc((size_t)D * D * 2);
  bf16* win_t  = (bf16*)alloc((size_t)F * D * 2);
  bf16* wout_t = (bf16*)alloc((size_t)D * F * 2);
  bf16* qbuf   = (bf16*)alloc((size_t)M * D * 2);
  bf16* kbuf   = (bf16*)alloc((size_t)M * D * 2);
  bf16* vtbuf  = (bf16*)alloc((size_t)M * D * 2);   // [B,1024(nh),2048(s)]
  bf16* zbuf   = (bf16*)alloc((size_t)M * D * 2);
  float* rmid  = (float*)alloc((size_t)M * D * 4);
  bf16* xn2    = (bf16*)alloc((size_t)M * D * 2);
  bf16* hbuf   = (bf16*)alloc((size_t)M * F * 2);

  dim3 tb(32, 8);

  // LN1
  ln_kernel<<<M, 256, 0, stream>>>(resid_pre, ln1_w, ln1_b, xn1, D);
  // Weight conversion / transposition (fp32 -> bf16 [N,K])
  transpose_head_w<<<dim3(2, 32, 16), tb, 0, stream>>>(W_Q, wq_t);
  transpose_head_w<<<dim3(2, 32, 16), tb, 0, stream>>>(W_K, wk_t);
  transpose_head_w<<<dim3(2, 32, 16), tb, 0, stream>>>(W_V, wv_t);
  transpose_f32_bf16<<<dim3(32, 32), tb, 0, stream>>>(W_O, wo_t, 1024, 1024);
  transpose_f32_bf16<<<dim3(128, 32), tb, 0, stream>>>(W_in, win_t, 1024, 4096);
  transpose_f32_bf16<<<dim3(32, 128), tb, 0, stream>>>(W_out, wout_t, 4096, 1024);

  // QKV projections (WMMA + async double-buffered LDS staging).
  gemm_kernel<0><<<dim3(D / 64, M / 256), 256, 0, stream>>>(
      xn1, wq_t, b_Q, nullptr, qbuf, M, D, D);
  gemm_kernel<0><<<dim3(D / 64, M / 256), 256, 0, stream>>>(
      xn1, wk_t, b_K, nullptr, kbuf, M, D, D);
  gemm_kernel<3><<<dim3(D / 64, M / 256), 256, 0, stream>>>(
      xn1, wv_t, b_V, nullptr, vtbuf, M, D, D);
  // Causal attention (WMMA + online softmax)
  attn_kernel<<<dim3(16, 16, 2), 256, 0, stream>>>(qbuf, kbuf, vtbuf, zbuf);
  // Output projection + residual (fp32 resid_mid)
  gemm_kernel<2><<<dim3(D / 64, M / 256), 256, 0, stream>>>(
      zbuf, wo_t, b_O, resid_pre, rmid, M, D, D);
  // LN2
  ln_kernel<<<M, 256, 0, stream>>>(rmid, ln2_w, ln2_b, xn2, D);
  // MLP in + exact GELU
  gemm_kernel<1><<<dim3(F / 64, M / 256), 256, 0, stream>>>(
      xn2, win_t, b_in, nullptr, hbuf, M, F, D);
  // MLP out + residual -> final output (fp32)
  gemm_kernel<2><<<dim3(D / 64, M / 256), 256, 0, stream>>>(
      hbuf, wout_t, b_out, rmid, out, M, D, F);
}
